// SWAttention_3908420239920
// MI455X (gfx1250) — matrix operand order repaired and checked by reading, not compiled, not run on hardware
//
#include <hip/hip_runtime.h>
#include <hip/hip_bf16.h>

// ---------------------------------------------------------------------------
// Swin-style shifted-window attention for MI455X (gfx1250, wave32, WMMA).
//   k1: roll(-3,-3) + x@w_qkv            (bf16 WMMA, f32 accum) -> qkv bf16
//   k2: per-(b,head,window) attention    (bf16 WMMA)            -> aout bf16
//   k3: aout@w_out + b_out + roll(+3,+3) (bf16 WMMA, f32 out)   -> d_out
// Grid order puts the N-tile index fastest so blocks sharing an A-slab run
// adjacently and hit MI455X's 192MB L2 instead of re-streaming HBM.
// Staging uses wide (b128) batched loads so global latency overlaps, and the
// already-bf16 tiles use GLOBAL_LOAD_ASYNC_TO_LDS (ASYNCcnt).
// ---------------------------------------------------------------------------

typedef __attribute__((ext_vector_type(16))) __bf16 v16bf;
typedef __attribute__((ext_vector_type(8)))  float  v8f;
typedef __bf16 bf16_t;

#define WSZ 7
#define HEADS 12
#define HEAD_DIM 32
#define DIM 384
#define INNER 384
#define QKV_N 1152
#define GRID 56
#define BATCH 32
#define NWIN 64            // 8x8 windows
#define NTOK 49            // tokens per window
#define NTOKP 64           // padded
#define M_TOT (BATCH * GRID * GRID)   // 100352
#define SCALE 0.17677669529663687f     // 1/sqrt(32)

// -------- optional CDNA5 async global->LDS staging (ASYNCcnt) --------------
#if defined(__HIP_DEVICE_COMPILE__) && \
    __has_builtin(__builtin_amdgcn_global_load_async_to_lds_b32) && \
    __has_builtin(__builtin_amdgcn_s_wait_asynccnt)
#define USE_ASYNC_LDS 1
#else
#define USE_ASYNC_LDS 0
#endif

typedef __attribute__((address_space(1))) int as1_int;
typedef __attribute__((address_space(3))) int as3_int;

// Copy 4 bytes global->LDS. Async path uses GLOBAL_LOAD_ASYNC_TO_LDS_B32.
// LDS AS3 offset = low 32 bits of the generic LDS address (ISA aperture rule).
__device__ __forceinline__ void copy_b32_g2l(const bf16_t* g, bf16_t* l) {
#if USE_ASYNC_LDS
    __builtin_amdgcn_global_load_async_to_lds_b32(
        (as1_int*)(size_t)g, (as3_int*)(unsigned int)(size_t)l, 0, 0);
#else
    *(unsigned int*)l = *(const unsigned int*)g;
#endif
}

__device__ __forceinline__ void async_wait_all() {
#if USE_ASYNC_LDS
    __builtin_amdgcn_s_wait_asynccnt(0);
#endif
}

union FragU { v16bf v; unsigned int u[8]; };
union U4x8 { uint4 q; bf16_t h[8]; };

// A operand 16x32 bf16 (ISA 7.12.2): lane half h, row = mBase + lane%16,
// VGPR j<4 -> K = 8h+2j, j>=4 -> K = 16+8h+2(j-4). K-pairs contiguous.
__device__ __forceinline__ v16bf load_frag_A(const bf16_t* lds, int stride,
                                             int mBase, int kOff) {
    const int lane = threadIdx.x & 31;
    const int half = lane >> 4;
    const bf16_t* row = lds + (mBase + (lane & 15)) * stride + kOff;
    FragU f;
#pragma unroll
    for (int j = 0; j < 8; ++j) {
        int kk = (j < 4) ? (half * 8 + 2 * j) : (16 + half * 8 + 2 * (j - 4));
        f.u[j] = *(const unsigned int*)(row + kk);
    }
    return f.v;
}

// B operand 32x16 bf16, LDS stored as [n][k] (k contiguous): lane half h,
// col n = nBase + lane%16, VGPR j<4 -> K = 8h+2j, j>=4 -> K = 16+8h+2(j-4).
__device__ __forceinline__ v16bf load_frag_B(const bf16_t* lds, int stride,
                                             int nBase, int kOff) {
    const int lane = threadIdx.x & 31;
    const int half = lane >> 4;
    const bf16_t* row = lds + (nBase + (lane & 15)) * stride + kOff;
    FragU f;
#pragma unroll
    for (int j = 0; j < 8; ++j)
        f.u[j] = *(const unsigned int*)(row + ((j < 4) ? (half * 8 + 2 * j) : (16 + half * 8 + 2 * (j - 4))));
    return f.v;
}

__device__ __forceinline__ v8f wmma_bf16(v16bf a, v16bf b, v8f c) {
    return __builtin_amdgcn_wmma_f32_16x16x32_bf16(false, a, false, b,
                                                   (short)0, c, false, false);
}

// C/D layout: VGPR r: lanes0-15 -> M=r, lanes16-31 -> M=8+r; N = lane%16.
__device__ __forceinline__ void store_tile_bf16(bf16_t* out, long ldc,
                                                int mb, int nb, v8f c) {
    const int lane = threadIdx.x & 31;
    const int n = nb + (lane & 15);
    const int m0 = mb + ((lane >> 4) ? 8 : 0);
#pragma unroll
    for (int r = 0; r < 8; ++r)
        out[(long)(m0 + r) * ldc + n] = (bf16_t)c[r];
}

// Stage a 32x64 fp32 weight slab w[kb+k][nBase+n] into LDS transposed [n][k]
// as bf16. Each thread: 4 x float4 batched loads, then scatter-convert.
__device__ __forceinline__ void stage_B_tile(const float* __restrict__ wgt,
                                             long ldw, int kb, int nBase,
                                             bf16_t* Bs) {
    const int t = threadIdx.x;
    const int nB = (t & 15) * 4;
    const int kB = t >> 4;                 // 0..7
    float4 b4[4];
#pragma unroll
    for (int i = 0; i < 4; ++i)
        b4[i] = *(const float4*)(wgt + (long)(kb + kB + i * 8) * ldw + nBase + nB);
#pragma unroll
    for (int i = 0; i < 4; ++i) {
        int k = kB + i * 8;
        Bs[(nB + 0) * 34 + k] = (bf16_t)b4[i].x;
        Bs[(nB + 1) * 34 + k] = (bf16_t)b4[i].y;
        Bs[(nB + 2) * 34 + k] = (bf16_t)b4[i].z;
        Bs[(nB + 3) * 34 + k] = (bf16_t)b4[i].w;
    }
}

// ---------------------------------------------------------------------------
// Kernel 1: qkv = roll(x) @ w_qkv     M=100352  K=384  N=1152
// grid = (N/64, M/64): N fastest -> A slab reused out of L2.
// ---------------------------------------------------------------------------
__global__ __launch_bounds__(128) void qkv_gemm_kernel(
        const float* __restrict__ x, const float* __restrict__ w_qkv,
        bf16_t* __restrict__ qkv) {
    __shared__ __align__(16) bf16_t As[64 * 34];
    __shared__ __align__(16) bf16_t Bs[64 * 34];
    const int nBase = blockIdx.x * 64;
    const int mBase = blockIdx.y * 64;
    const int t = threadIdx.x;
    const int wv = t >> 5, wm = wv >> 1, wn = wv & 1;
    v8f c00 = {}, c01 = {}, c10 = {}, c11 = {};

    // A staging map: thread owns 4 rows x 4 cols (float4). Row addresses of
    // the shifted-x rows are kb-invariant: precompute them once.
    const int c4 = (t & 7) * 4;            // col base 0,4,..,28
    const int r0 = t >> 3;                 // row base 0..15 (+16 per i)
    long rowAddr[4];
#pragma unroll
    for (int i = 0; i < 4; ++i) {
        int m = mBase + r0 + i * 16;
        int b_i = m / 3136;
        int rmd = m - b_i * 3136;
        int h = rmd / GRID, w = rmd - h * GRID;
        int hs = h + 3; if (hs >= GRID) hs -= GRID;
        int ws = w + 3; if (ws >= GRID) ws -= GRID;
        rowAddr[i] = (((long)b_i * GRID + hs) * GRID + ws) * DIM;
    }

    for (int kb = 0; kb < DIM; kb += 32) {
        // A tile: 4 batched float4 loads, then convert fp32 -> bf16
        float4 a4[4];
#pragma unroll
        for (int i = 0; i < 4; ++i)
            a4[i] = *(const float4*)(x + rowAddr[i] + kb + c4);
        if (kb + 32 < DIM)                        // uniform branch
            __builtin_prefetch(x + rowAddr[0] + kb + 32 + c4, 0, 3);
#pragma unroll
        for (int i = 0; i < 4; ++i) {
            bf16_t* dst = &As[(r0 + i * 16) * 34 + c4];
            dst[0] = (bf16_t)a4[i].x;
            dst[1] = (bf16_t)a4[i].y;
            dst[2] = (bf16_t)a4[i].z;
            dst[3] = (bf16_t)a4[i].w;
        }
        // B tile: w_qkv[k][n] -> LDS transposed [n][k]
        stage_B_tile(w_qkv, QKV_N, kb, nBase, Bs);
        __syncthreads();
        v16bf a0 = load_frag_A(As, 34, 32 * wm, 0);
        v16bf a1 = load_frag_A(As, 34, 32 * wm + 16, 0);
        v16bf b0 = load_frag_B(Bs, 34, 32 * wn, 0);
        v16bf b1 = load_frag_B(Bs, 34, 32 * wn + 16, 0);
        c00 = wmma_bf16(a0, b0, c00);
        c01 = wmma_bf16(a0, b1, c01);
        c10 = wmma_bf16(a1, b0, c10);
        c11 = wmma_bf16(a1, b1, c11);
        __syncthreads();
    }
    const int mb = mBase + 32 * wm, nb = nBase + 32 * wn;
    store_tile_bf16(qkv, QKV_N, mb,      nb,      c00);
    store_tile_bf16(qkv, QKV_N, mb,      nb + 16, c01);
    store_tile_bf16(qkv, QKV_N, mb + 16, nb,      c10);
    store_tile_bf16(qkv, QKV_N, mb + 16, nb + 16, c11);
}

// ---------------------------------------------------------------------------
// Kernel 2: per-(batch, head, window) attention. One block = 4 waves.
// ---------------------------------------------------------------------------
__global__ __launch_bounds__(128) void attn_kernel(
        const bf16_t* __restrict__ qkv, const float* __restrict__ pos_emb,
        bf16_t* __restrict__ aout) {
    __shared__ __align__(16) bf16_t Qs[NTOKP * 34];
    __shared__ __align__(16) bf16_t Ks[NTOKP * 34];
    __shared__ __align__(16) bf16_t Vt[HEAD_DIM * 66];   // transposed [dim][tok]
    __shared__ __align__(16) float  Sc[NTOKP * 65];
    __shared__ __align__(16) bf16_t At[NTOKP * 66];
    __shared__ float Pe[169];

    const int blk = blockIdx.x;
    const int b_i = blk / (HEADS * NWIN);
    const int rem = blk - b_i * (HEADS * NWIN);
    const int head = rem >> 6;
    const int win = rem & 63;
    const int wh = win >> 3, ww = win & 7;
    const int t = threadIdx.x;
    const int wv = t >> 5;

    for (int i = t; i < 169; i += 128) Pe[i] = pos_emb[i];

    // ---- stage Q,K (row-major [tok][dim], async copy: already bf16) -------
    // 49 rows x 16 dwords each
    for (int c = t; c < NTOK * 16; c += 128) {
        int tok = c >> 4, part = c & 15;
        int ti = tok / WSZ, tj = tok - ti * WSZ;
        long gb = (((long)b_i * GRID + wh * WSZ + ti) * GRID + ww * WSZ + tj)
                  * QKV_N + head * HEAD_DIM + part * 2;
        copy_b32_g2l(qkv + gb,         Qs + tok * 34 + part * 2);
        copy_b32_g2l(qkv + gb + INNER, Ks + tok * 34 + part * 2);
    }
    // zero-fill padded token rows 49..63 of Q/K
    for (int i = t; i < (NTOKP - NTOK) * 34; i += 128) {
        Qs[NTOK * 34 + i] = (bf16_t)0.0f;
        Ks[NTOK * 34 + i] = (bf16_t)0.0f;
    }
    // V staged transposed [dim][tok]: thread owns (token, 16-dim half),
    // two batched uint4 loads then b16 transpose-scatter into LDS.
    {
        int tok = t >> 1;                   // 0..63
        int dh = (t & 1) * 16;              // dim half base
        if (tok < NTOK) {
            int ti = tok / WSZ, tj = tok - ti * WSZ;
            long base = (((long)b_i * GRID + wh * WSZ + ti) * GRID
                         + ww * WSZ + tj) * QKV_N + head * HEAD_DIM
                        + 2 * INNER + dh;
            U4x8 u0, u1;
            u0.q = *(const uint4*)(qkv + base);
            u1.q = *(const uint4*)(qkv + base + 8);
#pragma unroll
            for (int i = 0; i < 8; ++i) {
                Vt[(dh + i) * 66 + tok]     = u0.h[i];
                Vt[(dh + 8 + i) * 66 + tok] = u1.h[i];
            }
        } else {
#pragma unroll
            for (int i = 0; i < 16; ++i)
                Vt[(dh + i) * 66 + tok] = (bf16_t)0.0f;
        }
    }
    async_wait_all();
    __syncthreads();

    // Scores: wave wv owns 16 rows x 64 cols; K = HEAD_DIM = 32 -> 1 WMMA/tile
    {
        v16bf a = load_frag_A(Qs, 34, 16 * wv, 0);
        const int lane = t & 31;
        const int n0 = lane & 15;
        const int m0 = 16 * wv + ((lane >> 4) ? 8 : 0);
#pragma unroll
        for (int tn = 0; tn < 4; ++tn) {
            v16bf b = load_frag_B(Ks, 34, 16 * tn, 0);
            v8f z = {};
            v8f c = wmma_bf16(a, b, z);
#pragma unroll
            for (int r = 0; r < 8; ++r)
                Sc[(m0 + r) * 65 + 16 * tn + n0] = c[r];
        }
    }
    __syncthreads();

    // Softmax with relative-position bias + shifted-window masks (1 thread/row)
    if (t < NTOKP) {
        const int i = t;
        if (i < NTOK) {
            const int xi = i / WSZ, yi = i - xi * WSZ;
            const bool lastRow = (wh == 7);   // last row of windows -> UL mask
            const bool lastCol = (ww == 7);   // last col of windows -> LR mask
            float mx = -1e30f;
            for (int j = 0; j < NTOKP; ++j) {
                float s = -1e30f;
                if (j < NTOK) {
                    int xj = j / WSZ, yj = j - xj * WSZ;
                    s = Sc[i * 65 + j] * SCALE
                      + Pe[(xj - xi + 6) * 13 + (yj - yi + 6)];
                    if (lastRow && ((i >= 28) != (j >= 28))) s = -1e30f;
                    if (lastCol && ((yi >= 4) != (yj >= 4))) s = -1e30f;
                }
                Sc[i * 65 + j] = s;
                mx = fmaxf(mx, s);
            }
            float sum = 0.0f;
            for (int j = 0; j < NTOKP; ++j) {
                float e = __expf(Sc[i * 65 + j] - mx);
                Sc[i * 65 + j] = e;
                sum += e;
            }
            float inv = 1.0f / sum;
            for (int j = 0; j < NTOKP; ++j)
                At[i * 66 + j] = (bf16_t)(Sc[i * 65 + j] * inv);
        } else {
            for (int j = 0; j < NTOKP; ++j) At[i * 66 + j] = (bf16_t)0.0f;
        }
    }
    __syncthreads();

    // attn @ V : K = 64 tokens (2 x 32), N = 32 dims (2 tiles)
    {
        v8f c0 = {}, c1 = {};
#pragma unroll
        for (int ko = 0; ko < NTOKP; ko += 32) {
            v16bf a  = load_frag_A(At, 66, 16 * wv, ko);
            v16bf b0 = load_frag_B(Vt, 66, 0,  ko);
            v16bf b1 = load_frag_B(Vt, 66, 16, ko);
            c0 = wmma_bf16(a, b0, c0);
            c1 = wmma_bf16(a, b1, c1);
        }
        const int lane = t & 31;
        const int n0 = lane & 15;
        const int m0 = 16 * wv + ((lane >> 4) ? 8 : 0);
#pragma unroll
        for (int r = 0; r < 8; ++r) {
            int tok = m0 + r;
            if (tok < NTOK) {
                int ti = tok / WSZ, tj = tok - ti * WSZ;
                int h = wh * WSZ + ti, w = ww * WSZ + tj;
                long base = (((long)b_i * GRID + h) * GRID + w) * INNER
                          + head * HEAD_DIM;
                aout[base + n0]      = (bf16_t)c0[r];
                aout[base + 16 + n0] = (bf16_t)c1[r];
            }
        }
    }
}

// ---------------------------------------------------------------------------
// Kernel 3: out = aout @ w_out + b_out, written with roll(+3,+3). N = 384.
// grid = (N/64, M/64): N fastest for L2 reuse of the A slab.
// ---------------------------------------------------------------------------
__global__ __launch_bounds__(128) void out_gemm_kernel(
        const bf16_t* __restrict__ ain, const float* __restrict__ w_out,
        const float* __restrict__ b_out, float* __restrict__ out) {
    __shared__ __align__(16) bf16_t As[64 * 34];
    __shared__ __align__(16) bf16_t Bs[64 * 34];
    const int nBase = blockIdx.x * 64;
    const int mBase = blockIdx.y * 64;
    const int t = threadIdx.x;
    const int wv = t >> 5, wm = wv >> 1, wn = wv & 1;
    v8f c00 = {}, c01 = {}, c10 = {}, c11 = {};

    for (int kb = 0; kb < INNER; kb += 32) {
        // A tile already bf16: async copy straight into LDS (64 rows x 16 dw)
        for (int c = t; c < 64 * 16; c += 128) {
            int row = c >> 4, part = c & 15;
            copy_b32_g2l(ain + (long)(mBase + row) * INNER + kb + part * 2,
                         As + row * 34 + part * 2);
        }
        stage_B_tile(w_out, DIM, kb, nBase, Bs);
        async_wait_all();
        __syncthreads();
        v16bf a0 = load_frag_A(As, 34, 32 * wm, 0);
        v16bf a1 = load_frag_A(As, 34, 32 * wm + 16, 0);
        v16bf b0 = load_frag_B(Bs, 34, 32 * wn, 0);
        v16bf b1 = load_frag_B(Bs, 34, 32 * wn + 16, 0);
        c00 = wmma_bf16(a0, b0, c00);
        c01 = wmma_bf16(a0, b1, c01);
        c10 = wmma_bf16(a1, b0, c10);
        c11 = wmma_bf16(a1, b1, c11);
        __syncthreads();
    }

    const int lane = t & 31;
    const int half = lane >> 4;
#pragma unroll
    for (int tile = 0; tile < 4; ++tile) {
        const int tm = tile >> 1, tn = tile & 1;
        v8f c = (tile == 0) ? c00 : (tile == 1) ? c01 : (tile == 2) ? c10 : c11;
        const int n = nBase + 32 * wn + 16 * tn + (lane & 15);
        const int m0 = mBase + 32 * wm + 16 * tm + half * 8;
        const float bias = b_out[n];
#pragma unroll
        for (int r = 0; r < 8; ++r) {
            int m = m0 + r;
            int b_i = m / 3136;
            int rmd = m - b_i * 3136;
            int h = rmd / GRID, w = rmd - h * GRID;
            int hd = h + 3; if (hd >= GRID) hd -= GRID;
            int wd = w + 3; if (wd >= GRID) wd -= GRID;
            out[(((long)b_i * GRID + hd) * GRID + wd) * DIM + n] = c[r] + bias;
        }
    }
}

// ---------------------------------------------------------------------------
extern "C" void kernel_launch(void* const* d_in, const int* in_sizes, int n_in,
                              void* d_out, int out_size, void* d_ws, size_t ws_size,
                              hipStream_t stream) {
    const float* x       = (const float*)d_in[0];
    const float* w_qkv   = (const float*)d_in[1];
    const float* pos_emb = (const float*)d_in[2];
    const float* w_out   = (const float*)d_in[3];
    const float* b_out   = (const float*)d_in[4];
    float* out = (float*)d_out;

    // workspace: qkv bf16 [100352 x 1152] then attn-out bf16 [100352 x 384]
    bf16_t* qkv  = (bf16_t*)d_ws;
    bf16_t* aout = (bf16_t*)((char*)d_ws + (size_t)M_TOT * QKV_N * 2);

    qkv_gemm_kernel<<<dim3(QKV_N / 64, M_TOT / 64), 128, 0, stream>>>(x, w_qkv, qkv);
    attn_kernel<<<BATCH * HEADS * NWIN, 128, 0, stream>>>(qkv, pos_emb, aout);
    out_gemm_kernel<<<dim3(DIM / 64, M_TOT / 64), 128, 0, stream>>>(aout, w_out, b_out, out);
}
